// RecCellEncoder_3599182594229
// MI455X (gfx1250) — matrix operand order repaired and checked by reading, not compiled, not run on hardware
//
#include <hip/hip_runtime.h>
#include <hip/hip_bf16.h>
#include <stdint.h>

// ---------------------------------------------------------------------------
// Bidirectional GRU encoder for MI455X (gfx1250, wave32, WMMA bf16 path).
// B=128, T=256, D=1024, H=1024.
//  Phase 1: cast x / Wi / Wh to bf16, zero hidden state.
//  Phase 2: Gi = X @ Wi^T (both dirs) -- one large WMMA GEMM, bf16 output.
//  Phase 3: 256 sequential steps; ONE fused kernel per step computes
//           gh = h @ Wh^T for all 3 gate strips of a j-block with WMMA and
//           applies sigmoid/tanh/mask update entirely in registers.
// ---------------------------------------------------------------------------

#define B_  128
#define T_  256
#define D_  1024
#define H_  1024
#define G3H (3 * H_)

typedef __attribute__((ext_vector_type(16))) __bf16 v16bf;
typedef __attribute__((ext_vector_type(8)))  float  v8f;

static __device__ __forceinline__ unsigned short f2bf(float f) {
    union { float f; uint32_t u; } v; v.f = f;
    uint32_t r = v.u + 0x7FFFu + ((v.u >> 16) & 1u);   // round-to-nearest-even
    return (unsigned short)(r >> 16);
}
static __device__ __forceinline__ float bf2f(unsigned short s) {
    union { uint32_t u; float f; } v; v.u = ((uint32_t)s) << 16;
    return v.f;
}

// ---------------------------------------------------------------------------
// fp32 -> bf16 cast kernel
// ---------------------------------------------------------------------------
__global__ void cast_f32_to_bf16(const float* __restrict__ in,
                                 unsigned short* __restrict__ out, int n) {
    int i = blockIdx.x * blockDim.x + threadIdx.x;
    if (i < n) out[i] = f2bf(in[i]);
}

__global__ void zero_state(float* __restrict__ hf,
                           unsigned short* __restrict__ hb, int n) {
    int i = blockIdx.x * blockDim.x + threadIdx.x;
    if (i < n) { hf[i] = 0.0f; hb[i] = 0; }
}

// ---------------------------------------------------------------------------
// Generic bf16 WMMA GEMM (used for the big input projection):
//   C[M,N] = A[M,K] * B[N,K]^T, B in native weight layout [N][K].
// Block tile 128x128, K staged 32-deep through LDS, 8 waves, each wave
// 2x4 16x16 tiles -> 8 v_wmma_f32_16x16x32_bf16 per K-slice.
// ---------------------------------------------------------------------------
#define BLK_M 128
#define BLK_N 128
#define BLK_K 32
#define LDAp  40   // padded LDS row (elements); 80B rows, 16B-aligned chunks

union FragAB { v16bf v; uint4 q[2]; };
union FragC  { v8f  v; float f[8]; };

template <typename OutT>
static __device__ __forceinline__ void store_out(OutT* p, float v);
template <>
__device__ __forceinline__ void store_out<unsigned short>(unsigned short* p, float v) { *p = f2bf(v); }
template <>
__device__ __forceinline__ void store_out<float>(float* p, float v) { *p = v; }

template <typename OutT>
__global__ __launch_bounds__(256)
void gemm_bf16_wmma(const unsigned short* __restrict__ Abase,
                    const unsigned short* __restrict__ Bbase,
                    OutT* __restrict__ Cbase,
                    int M, int N, int K,
                    size_t strideA, size_t strideB, size_t strideC) {
    __shared__ unsigned short As[BLK_M * LDAp];
    __shared__ unsigned short Bs[BLK_N * LDAp];

    const unsigned short* A  = Abase + (size_t)blockIdx.z * strideA;
    const unsigned short* Bw = Bbase + (size_t)blockIdx.z * strideB;
    OutT*                 C  = Cbase + (size_t)blockIdx.z * strideC;

    const int tid  = threadIdx.x;
    const int wave = tid >> 5;
    const int lane = tid & 31;
    const int wm   = wave & 3;   // 4 waves along M, 32 rows each
    const int wn   = wave >> 2;  // 2 waves along N, 64 cols each
    const int rowbase = blockIdx.y * BLK_M;
    const int colbase = blockIdx.x * BLK_N;

    const int srow = tid >> 1;         // 0..127
    const int skp  = (tid & 1) * 16;   // 0 or 16 (elements)

    v8f acc[2][4];
#pragma unroll
    for (int i = 0; i < 2; ++i)
#pragma unroll
        for (int j = 0; j < 4; ++j)
#pragma unroll
            for (int e = 0; e < 8; ++e) acc[i][j][e] = 0.0f;

    const int lrow  = lane & 15;
    const int half8 = (lane < 16) ? 0 : 8;    // A fragment K-group base
    const int bK    = (lane < 16) ? 0 : 16;   // unused; B fragment also uses half8

    const int kblocks = K >> 5;
    for (int kb = 0; kb < kblocks; ++kb) {
        const int kg = kb * BLK_K + skp;
        const uint4* gA = (const uint4*)(A  + (size_t)(rowbase + srow) * K + kg);
        const uint4* gB = (const uint4*)(Bw + (size_t)(colbase + srow) * K + kg);
        uint4 a0 = gA[0], a1 = gA[1];
        uint4 b0 = gB[0], b1 = gB[1];

        __syncthreads();
        *(uint4*)&As[srow * LDAp + skp]     = a0;
        *(uint4*)&As[srow * LDAp + skp + 8] = a1;
        *(uint4*)&Bs[srow * LDAp + skp]     = b0;
        *(uint4*)&Bs[srow * LDAp + skp + 8] = b1;
        __syncthreads();

        FragAB af[2];
#pragma unroll
        for (int tm = 0; tm < 2; ++tm) {
            const unsigned short* p = &As[(wm * 32 + tm * 16 + lrow) * LDAp + half8];
            af[tm].q[0] = *(const uint4*)p;
            af[tm].q[1] = *(const uint4*)(p + 16);
        }
        FragAB bfr[4];
#pragma unroll
        for (int tn = 0; tn < 4; ++tn) {
            const unsigned short* p = &Bs[(wn * 64 + tn * 16 + lrow) * LDAp + half8];
            bfr[tn].q[0] = *(const uint4*)p;
            bfr[tn].q[1] = *(const uint4*)(p + 16);
        }
#pragma unroll
        for (int tm = 0; tm < 2; ++tm)
#pragma unroll
            for (int tn = 0; tn < 4; ++tn)
                acc[tm][tn] = __builtin_amdgcn_wmma_f32_16x16x32_bf16(
                    false, af[tm].v, false, bfr[tn].v,
                    (short)0, acc[tm][tn], false, false);
    }

    const int mhalf = (lane < 16) ? 0 : 8;
#pragma unroll
    for (int tm = 0; tm < 2; ++tm) {
#pragma unroll
        for (int tn = 0; tn < 4; ++tn) {
            FragC c; c.v = acc[tm][tn];
            const int ncol = colbase + wn * 64 + tn * 16 + lrow;
#pragma unroll
            for (int r = 0; r < 8; ++r) {
                const int mrow = rowbase + wm * 32 + tm * 16 + mhalf + r;
                store_out<OutT>(C + (size_t)mrow * N + ncol, c.f[r]);
            }
        }
    }
}

// ---------------------------------------------------------------------------
// Fused recurrent step: one launch per time step, both directions (grid.z).
// Each block owns a 64-wide j-strip and computes gh for ALL THREE gate strips
// (r,z,n) of that strip with WMMA, then performs the full GRU update in
// registers (the lane holding r[b][j] also holds z[b][j], n[b][j] because all
// three accumulator sets share the C-matrix layout). No gh memory round-trip.
// Waves: 4 along M (batch, 32 rows) x 2 along N (j, 32 cols); per wave
// 3 gates x 2x2 tiles = 12 wmma per 32-deep K slice.
// ---------------------------------------------------------------------------
__global__ __launch_bounds__(256)
void gru_step_fused(const unsigned short* __restrict__ hBms,  // [2][B][H] bf16
                    const unsigned short* __restrict__ Whb,   // [2][3H][H] bf16
                    const unsigned short* __restrict__ Gi,    // [2][B*T][3H] bf16
                    const float* __restrict__ bi_f, const float* __restrict__ bh_f,
                    const float* __restrict__ bi_r, const float* __restrict__ bh_r,
                    const float* __restrict__ mask,           // [B][T]
                    float* __restrict__ h_f32,                // [2][B][H]
                    unsigned short* __restrict__ h_bf,        // [2][B][H]
                    float* __restrict__ out,                  // [B][T][2H]
                    int t) {
    __shared__ unsigned short As[B_ * LDAp];        // 128 x 32 (padded)
    __shared__ unsigned short Bs[192 * LDAp];       // 3 gates x 64 j x 32 k

    const int dir   = blockIdx.z;
    const int ti    = dir ? (T_ - 1 - t) : t;
    const int jbase = blockIdx.x * 64;

    const unsigned short* A  = hBms + (size_t)dir * (B_ * H_);
    const unsigned short* Bw = Whb  + (size_t)dir * ((size_t)G3H * H_);
    const float* bi = dir ? bi_r : bi_f;
    const float* bh = dir ? bh_r : bh_f;

    const int tid  = threadIdx.x;
    const int wave = tid >> 5;
    const int lane = tid & 31;
    const int wm   = wave & 3;   // batch rows: 32 each
    const int wn   = wave >> 2;  // j cols: 32 each

    v8f acc[3][2][2];
#pragma unroll
    for (int g = 0; g < 3; ++g)
#pragma unroll
        for (int i = 0; i < 2; ++i)
#pragma unroll
            for (int j = 0; j < 2; ++j)
#pragma unroll
                for (int e = 0; e < 8; ++e) acc[g][i][j][e] = 0.0f;

    const int lrow  = lane & 15;
    const int half8 = (lane < 16) ? 0 : 8;
    const int bK    = (lane < 16) ? 0 : 16;

    for (int kb = 0; kb < (H_ >> 5); ++kb) {
        const int k0 = kb * 32;
        __syncthreads();
        // Stage A (512 8-elem chunks) + 3 gate strips of Wh (768 chunks):
        // 1280 chunks / 256 threads = 5 fully-active iterations.
#pragma unroll
        for (int it = 0; it < 5; ++it) {
            const int c = tid + it * 256;
            const unsigned short* src;
            unsigned short* dst;
            if (c < 512) {
                const int row = c >> 2, kp = (c & 3) * 8;
                src = A + (size_t)row * H_ + k0 + kp;
                dst = &As[row * LDAp + kp];
            } else {
                const int cb = c - 512;
                const int row = cb >> 2, kp = (cb & 3) * 8;   // row: 0..191
                const int g = row >> 6, n = row & 63;
                src = Bw + (size_t)(g * H_ + jbase + n) * H_ + k0 + kp;
                dst = &Bs[row * LDAp + kp];
            }
            *(uint4*)dst = *(const uint4*)src;
        }
        __syncthreads();

        FragAB af[2];
#pragma unroll
        for (int tm = 0; tm < 2; ++tm) {
            const unsigned short* p = &As[(wm * 32 + tm * 16 + lrow) * LDAp + half8];
            af[tm].q[0] = *(const uint4*)p;
            af[tm].q[1] = *(const uint4*)(p + 16);
        }
#pragma unroll
        for (int g = 0; g < 3; ++g) {
            FragAB bfr[2];
#pragma unroll
            for (int tn = 0; tn < 2; ++tn) {
                const unsigned short* p = &Bs[(g * 64 + wn * 32 + tn * 16 + lrow) * LDAp + half8];
                bfr[tn].q[0] = *(const uint4*)p;
                bfr[tn].q[1] = *(const uint4*)(p + 16);
            }
#pragma unroll
            for (int tm = 0; tm < 2; ++tm)
#pragma unroll
                for (int tn = 0; tn < 2; ++tn)
                    acc[g][tm][tn] = __builtin_amdgcn_wmma_f32_16x16x32_bf16(
                        false, af[tm].v, false, bfr[tn].v,
                        (short)0, acc[g][tm][tn], false, false);
        }
    }

    // ---- fused GRU update, entirely from registers ----
    const int mhalf = (lane < 16) ? 0 : 8;
    const size_t gibase = (size_t)dir * (size_t)(B_ * T_) * G3H;
    const size_t hbase  = (size_t)dir * (size_t)(B_ * H_);
#pragma unroll
    for (int tn = 0; tn < 2; ++tn) {
        const int j = jbase + wn * 32 + tn * 16 + lrow;
        const float bir = bi[j], biz = bi[H_ + j], bin = bi[2 * H_ + j];
        const float bhr = bh[j], bhz = bh[H_ + j], bhn = bh[2 * H_ + j];
#pragma unroll
        for (int tm = 0; tm < 2; ++tm) {
            FragC cr, cz, cn;
            cr.v = acc[0][tm][tn];
            cz.v = acc[1][tm][tn];
            cn.v = acc[2][tm][tn];
#pragma unroll
            for (int r8 = 0; r8 < 8; ++r8) {
                const int b = wm * 32 + tm * 16 + mhalf + r8;
                const size_t girow = gibase + (size_t)(b * T_ + ti) * G3H;
                const float i_r = bf2f(Gi[girow +          j]) + bir;
                const float i_z = bf2f(Gi[girow +   H_ +   j]) + biz;
                const float i_n = bf2f(Gi[girow + 2*H_ +   j]) + bin;
                const float rg  = 1.0f / (1.0f + expf(-(i_r + cr.f[r8] + bhr)));
                const float zg  = 1.0f / (1.0f + expf(-(i_z + cz.f[r8] + bhz)));
                const float ng  = tanhf(i_n + rg * (cn.f[r8] + bhn));
                const size_t hoff = hbase + (size_t)b * H_ + j;
                const float hp = h_f32[hoff];
                const float hnew = (1.0f - zg) * ng + zg * hp;
                const float m  = mask[b * T_ + ti];
                const float ho = m * hnew + (1.0f - m) * hp;
                h_f32[hoff] = ho;
                h_bf[hoff]  = f2bf(ho);
                out[(size_t)(b * T_ + ti) * (2 * H_) + dir * H_ + j] = ho;
            }
        }
    }
}

// ---------------------------------------------------------------------------
// Launcher
// ---------------------------------------------------------------------------
extern "C" void kernel_launch(void* const* d_in, const int* in_sizes, int n_in,
                              void* d_out, int out_size, void* d_ws, size_t ws_size,
                              hipStream_t stream) {
    const float* x    = (const float*)d_in[0];
    const float* mask = (const float*)d_in[1];
    const float* Wi_f = (const float*)d_in[2];
    const float* Wh_f = (const float*)d_in[3];
    const float* bi_f = (const float*)d_in[4];
    const float* bh_f = (const float*)d_in[5];
    const float* Wi_r = (const float*)d_in[6];
    const float* Wh_r = (const float*)d_in[7];
    const float* bi_r = (const float*)d_in[8];
    const float* bh_r = (const float*)d_in[9];
    float* out = (float*)d_out;

    // ---- workspace layout (bytes) ----
    char* ws = (char*)d_ws;
    constexpr size_t XB_BYTES  = (size_t)B_ * T_ * D_ * 2;        //  64 MiB bf16 x
    constexpr size_t WIB_BYTES = (size_t)2 * G3H * D_ * 2;        //  12 MiB bf16 Wi f+r
    constexpr size_t WHB_BYTES = (size_t)2 * G3H * H_ * 2;        //  12 MiB bf16 Wh f+r
    constexpr size_t GI_BYTES  = (size_t)2 * B_ * T_ * G3H * 2;   // 384 MiB bf16 gi f+r
    constexpr size_t HF_BYTES  = (size_t)2 * B_ * H_ * 4;         //   1 MiB fp32 h
    size_t off = 0;
    unsigned short* Xb  = (unsigned short*)(ws + off); off += XB_BYTES;
    unsigned short* Wib = (unsigned short*)(ws + off); off += WIB_BYTES;
    unsigned short* Whb = (unsigned short*)(ws + off); off += WHB_BYTES;
    unsigned short* Gi  = (unsigned short*)(ws + off); off += GI_BYTES;
    float*          hF  = (float*)         (ws + off); off += HF_BYTES;
    unsigned short* hB  = (unsigned short*)(ws + off);

    // ---- 1) casts to bf16 + state init ----
    {
        int n = B_ * T_ * D_;
        cast_f32_to_bf16<<<(n + 255) / 256, 256, 0, stream>>>(x, Xb, n);
        int nw = G3H * D_;
        cast_f32_to_bf16<<<(nw + 255) / 256, 256, 0, stream>>>(Wi_f, Wib, nw);
        cast_f32_to_bf16<<<(nw + 255) / 256, 256, 0, stream>>>(Wi_r, Wib + (size_t)G3H * D_, nw);
        int nh = G3H * H_;
        cast_f32_to_bf16<<<(nh + 255) / 256, 256, 0, stream>>>(Wh_f, Whb, nh);
        cast_f32_to_bf16<<<(nh + 255) / 256, 256, 0, stream>>>(Wh_r, Whb + (size_t)G3H * H_, nh);
        int ns = 2 * B_ * H_;
        zero_state<<<(ns + 255) / 256, 256, 0, stream>>>(hF, hB, ns);
    }

    // ---- 2) big input-projection GEMM: Gi = X @ Wi^T (both directions) ----
    {
        dim3 grid(G3H / BLK_N, (B_ * T_) / BLK_M, 2);
        gemm_bf16_wmma<unsigned short><<<grid, 256, 0, stream>>>(
            Xb, Wib, Gi,
            B_ * T_, G3H, D_,
            /*strideA=*/0,
            /*strideB=*/(size_t)G3H * D_,
            /*strideC=*/(size_t)B_ * T_ * G3H);
    }

    // ---- 3) sequential recurrence: ONE fused launch per step ----
    for (int t = 0; t < T_; ++t) {
        dim3 grid(H_ / 64, 1, 2);
        gru_step_fused<<<grid, 256, 0, stream>>>(
            hB, Whb, Gi, bi_f, bh_f, bi_r, bh_r, mask, hF, hB, out, t);
    }
}
